// RGCNModel_87299505258751
// MI455X (gfx1250) — matrix operand order repaired and checked by reading, not compiled, not run on hardware
//
#include <hip/hip_runtime.h>
#include <hip/hip_bf16.h>

// RGCN 2-layer forward, transform-then-aggregate formulation.
// GEMMs use v_wmma_f32_16x16x32_bf16; scatters use global f32 atomics.

typedef __attribute__((ext_vector_type(16))) __bf16 v16bf;
typedef __attribute__((ext_vector_type(8)))  __bf16 v8bf;
typedef __attribute__((ext_vector_type(8)))  float  v8f;

#define RNUM 4
#define INF  384
#define HIDF 256
#define OUTF 64
#define NC1  1280   // 4*HID + HID (root)
#define NC2  320    // 4*OUT + OUT (root)

__global__ void fill_zero_u32(unsigned int* __restrict__ p, int n) {
    int i = blockIdx.x * blockDim.x + threadIdx.x;
    if (i < n) p[i] = 0u;
}

__global__ void count_edges(const int* __restrict__ dst, const int* __restrict__ et,
                            unsigned int* __restrict__ cnt, int E) {
    int e = blockIdx.x * blockDim.x + threadIdx.x;
    if (e < E) atomicAdd(&cnt[dst[e] * RNUM + et[e]], 1u);
}

__global__ void inv_counts(const unsigned int* __restrict__ cnt, float* __restrict__ inv, int n) {
    int i = blockIdx.x * blockDim.x + threadIdx.x;
    if (i < n) {
        unsigned int c = cnt[i];
        inv[i] = 1.0f / (float)(c > 1u ? c : 1u);
    }
}

__global__ void cvt_f32_bf16(const float* __restrict__ in, __bf16* __restrict__ out, long long n) {
    long long i = (long long)blockIdx.x * blockDim.x + threadIdx.x;
    if (i < n) out[i] = (__bf16)in[i];
}

// Wt[col*INF + k]: cols [0,1024) = W1[r][k][c] (r=col/256,c=col%256); cols [1024,1280) = root1[k][col-1024]
__global__ void build_w1t(const float* __restrict__ W1, const float* __restrict__ root1,
                          __bf16* __restrict__ Wt) {
    int idx = blockIdx.x * blockDim.x + threadIdx.x;
    if (idx >= NC1 * INF) return;
    int col = idx / INF, k = idx % INF;
    float v;
    if (col < 4 * HIDF) {
        int r = col >> 8, c = col & 255;
        v = W1[((long long)r * INF + k) * HIDF + c];
    } else {
        v = root1[(long long)k * HIDF + (col - 4 * HIDF)];
    }
    Wt[idx] = (__bf16)v;
}

// Wt[col*HIDF + k]: cols [0,256) = W2[r][k][c] (r=col/64,c=col%64); cols [256,320) = root2[k][col-256]
__global__ void build_w2t(const float* __restrict__ W2, const float* __restrict__ root2,
                          __bf16* __restrict__ Wt) {
    int idx = blockIdx.x * blockDim.x + threadIdx.x;
    if (idx >= NC2 * HIDF) return;
    int col = idx / HIDF, k = idx % HIDF;
    float v;
    if (col < 4 * OUTF) {
        int r = col >> 6, c = col & 63;
        v = W2[((long long)r * HIDF + k) * OUTF + c];
    } else {
        v = root2[(long long)k * OUTF + (col - 4 * OUTF)];
    }
    Wt[idx] = (__bf16)v;
}

// C[M,ND] (bf16) = A[M,KD] (bf16, row-major) x Bt[ND,KD]^T (bf16, col panels contiguous in k)
// Block: 128 threads = 4 waves; tile 64x64; each wave 32x32 (2x2 WMMA 16x16x32 tiles).
template <int KD, int ND>
__global__ void __launch_bounds__(128) gemm_bf16(const __bf16* __restrict__ A,
                                                 const __bf16* __restrict__ Bt,
                                                 __bf16* __restrict__ C, int M) {
    const int lane = threadIdx.x & 31;
    const int wave = threadIdx.x >> 5;
    const int n_base = blockIdx.x * 64 + (wave & 1) * 32;
    const int m_base = blockIdx.y * 64 + (wave >> 1) * 32;
    if (m_base >= M) return;
    const int lrow = lane & 15;   // row-in-tile (A), col-in-tile (B/C)
    const int lhi  = lane >> 4;   // K-half select (A), K-16-group (B), M+8 (C)

    v8f acc[2][2] = {};

    int r0 = m_base + lrow;       if (r0 >= M) r0 = M - 1;
    int r1 = m_base + 16 + lrow;  if (r1 >= M) r1 = M - 1;
    const __bf16* a0p = A + (long long)r0 * KD + lhi * 16;
    const __bf16* a1p = A + (long long)r1 * KD + lhi * 16;
    const __bf16* b0p = Bt + (long long)(n_base + lrow) * KD + lhi * 16;
    const __bf16* b1p = Bt + (long long)(n_base + 16 + lrow) * KD + lhi * 16;

#pragma unroll
    for (int kk = 0; kk < KD; kk += 32) {
        v16bf a[2], b[2];
        {
            v8bf lo = *(const v8bf*)(a0p + kk);
            v8bf hi = *(const v8bf*)(a0p + kk + 8);
#pragma unroll
            for (int t = 0; t < 8; t++) { a[0][t] = lo[t]; a[0][t + 8] = hi[t]; }
        }
        {
            v8bf lo = *(const v8bf*)(a1p + kk);
            v8bf hi = *(const v8bf*)(a1p + kk + 8);
#pragma unroll
            for (int t = 0; t < 8; t++) { a[1][t] = lo[t]; a[1][t + 8] = hi[t]; }
        }
        b[0] = *(const v16bf*)(b0p + kk);
        b[1] = *(const v16bf*)(b1p + kk);
#pragma unroll
        for (int i = 0; i < 2; i++)
#pragma unroll
            for (int j = 0; j < 2; j++)
                acc[i][j] = __builtin_amdgcn_wmma_f32_16x16x32_bf16(
                    false, a[i], false, b[j], (short)0, acc[i][j], false, false);
    }

#pragma unroll
    for (int i = 0; i < 2; i++) {
        int m0 = m_base + i * 16;
        if (m0 >= M) continue;
#pragma unroll
        for (int j = 0; j < 2; j++) {
            int c0 = n_base + j * 16 + lrow;
            int rb = m0 + lhi * 8;
#pragma unroll
            for (int v = 0; v < 8; v++) {
                int row = rb + v;
                if (row < M) C[(long long)row * ND + c0] = (__bf16)acc[i][j][v];
            }
        }
    }
}

// hacc[n,h] = y1[n, 1024+h] + b1[h]   (root + bias)
__global__ void init_h(const __bf16* __restrict__ y1, const float* __restrict__ b1,
                       float* __restrict__ hacc, int total) {
    int i = blockIdx.x * blockDim.x + threadIdx.x;
    if (i >= total) return;
    int n = i >> 8, h = i & 255;
    hacc[i] = (float)y1[(long long)n * NC1 + 4 * HIDF + h] + b1[h];
}

// one wave per edge; lane handles 8 consecutive h
__global__ void scatter1(const int* __restrict__ src, const int* __restrict__ dst,
                         const int* __restrict__ et, const __bf16* __restrict__ y1,
                         const float* __restrict__ inv, float* __restrict__ hacc, int E) {
    int wave = (int)((blockIdx.x * blockDim.x + threadIdx.x) >> 5);
    int lane = threadIdx.x & 31;
    if (wave >= E) return;
    int s = src[wave], d = dst[wave], r = et[wave];
    float w = inv[d * RNUM + r];
    const __bf16* yp = y1 + (long long)s * NC1 + r * HIDF + lane * 8;
    v8bf v = *(const v8bf*)yp;
    float* hp = hacc + (long long)d * HIDF + lane * 8;
#pragma unroll
    for (int t = 0; t < 8; t++) atomicAdd(hp + t, (float)v[t] * w);
}

__global__ void bn_relu_bf16(const float* __restrict__ hacc,
                             const float* __restrict__ gamma, const float* __restrict__ beta,
                             const float* __restrict__ mean, const float* __restrict__ var,
                             __bf16* __restrict__ hb, int total) {
    int i = blockIdx.x * blockDim.x + threadIdx.x;
    if (i >= total) return;
    int h = i & 255;
    float v = hacc[i];
    v = (v - mean[h]) * rsqrtf(var[h] + 1e-5f) * gamma[h] + beta[h];
    hb[i] = (__bf16)fmaxf(v, 0.0f);
}

// out[n,o] = y2[n, 256+o] + b2[o]
__global__ void init_out(const __bf16* __restrict__ y2, const float* __restrict__ b2,
                         float* __restrict__ out, int total) {
    int i = blockIdx.x * blockDim.x + threadIdx.x;
    if (i >= total) return;
    int n = i >> 6, o = i & 63;
    out[i] = (float)y2[(long long)n * NC2 + 4 * OUTF + o] + b2[o];
}

// one wave per edge; lane handles o = lane and lane+32
__global__ void scatter2(const int* __restrict__ src, const int* __restrict__ dst,
                         const int* __restrict__ et, const __bf16* __restrict__ y2,
                         const float* __restrict__ inv, float* __restrict__ out, int E) {
    int wave = (int)((blockIdx.x * blockDim.x + threadIdx.x) >> 5);
    int lane = threadIdx.x & 31;
    if (wave >= E) return;
    int s = src[wave], d = dst[wave], r = et[wave];
    float w = inv[d * RNUM + r];
    const __bf16* yp = y2 + (long long)s * NC2 + r * OUTF;
    float* op = out + (long long)d * OUTF;
    atomicAdd(op + lane,      (float)yp[lane] * w);
    atomicAdd(op + lane + 32, (float)yp[lane + 32] * w);
}

extern "C" void kernel_launch(void* const* d_in, const int* in_sizes, int n_in,
                              void* d_out, int out_size, void* d_ws, size_t ws_size,
                              hipStream_t stream) {
    const float* x      = (const float*)d_in[0];
    const int*   eidx   = (const int*)d_in[1];
    const int*   etype  = (const int*)d_in[2];
    const float* W1     = (const float*)d_in[3];
    const float* root1  = (const float*)d_in[4];
    const float* b1     = (const float*)d_in[5];
    const float* bng    = (const float*)d_in[6];
    const float* bnb    = (const float*)d_in[7];
    const float* bnm    = (const float*)d_in[8];
    const float* bnv    = (const float*)d_in[9];
    const float* W2     = (const float*)d_in[10];
    const float* root2  = (const float*)d_in[11];
    const float* b2     = (const float*)d_in[12];
    float* out = (float*)d_out;

    const int N = in_sizes[0] / INF;
    const int E = in_sizes[2];
    const int* esrc = eidx;
    const int* edst = eidx + E;

    // workspace carve-up (256B aligned)
    char* ws = (char*)d_ws;
    size_t off = 0;
    auto carve = [&](size_t bytes) -> void* {
        void* p = ws + off;
        off = (off + bytes + 255) & ~(size_t)255;
        return p;
    };
    unsigned int* cnt = (unsigned int*)carve((size_t)N * RNUM * 4);
    float*  inv  = (float*)carve((size_t)N * RNUM * 4);
    __bf16* xb   = (__bf16*)carve((size_t)N * INF * 2);
    __bf16* w1t  = (__bf16*)carve((size_t)NC1 * INF * 2);
    __bf16* y1   = (__bf16*)carve((size_t)N * NC1 * 2);
    float*  hacc = (float*)carve((size_t)N * HIDF * 4);
    __bf16* hb   = (__bf16*)carve((size_t)N * HIDF * 2);
    __bf16* w2t  = (__bf16*)carve((size_t)NC2 * HIDF * 2);
    __bf16* y2   = (__bf16*)carve((size_t)N * NC2 * 2);
    (void)ws_size;

    const int B = 256;
    auto blocks = [](long long n, int b) { return (unsigned)((n + b - 1) / b); };

    // degree counts -> inverse means
    fill_zero_u32<<<blocks((long long)N * RNUM, B), B, 0, stream>>>(cnt, N * RNUM);
    count_edges<<<blocks(E, B), B, 0, stream>>>(edst, etype, cnt, E);
    inv_counts<<<blocks((long long)N * RNUM, B), B, 0, stream>>>(cnt, inv, N * RNUM);

    // bf16 staging
    cvt_f32_bf16<<<blocks((long long)N * INF, B), B, 0, stream>>>(x, xb, (long long)N * INF);
    build_w1t<<<blocks((long long)NC1 * INF, B), B, 0, stream>>>(W1, root1, w1t);

    // layer 1: transform, aggregate, BN+ReLU
    {
        dim3 grid(NC1 / 64, (unsigned)((N + 63) / 64));
        gemm_bf16<INF, NC1><<<grid, 128, 0, stream>>>(xb, w1t, y1, N);
    }
    init_h<<<blocks((long long)N * HIDF, B), B, 0, stream>>>(y1, b1, hacc, N * HIDF);
    scatter1<<<blocks((long long)E * 32, B), B, 0, stream>>>(esrc, edst, etype, y1, inv, hacc, E);
    bn_relu_bf16<<<blocks((long long)N * HIDF, B), B, 0, stream>>>(hacc, bng, bnb, bnm, bnv, hb, N * HIDF);

    // layer 2: transform, aggregate
    build_w2t<<<blocks((long long)NC2 * HIDF, B), B, 0, stream>>>(W2, root2, w2t);
    {
        dim3 grid(NC2 / 64, (unsigned)((N + 63) / 64));
        gemm_bf16<HIDF, NC2><<<grid, 128, 0, stream>>>(hb, w2t, y2, N);
    }
    init_out<<<blocks((long long)N * OUTF, B), B, 0, stream>>>(y2, b2, out, N * OUTF);
    scatter2<<<blocks((long long)E * 32, B), B, 0, stream>>>(esrc, edst, etype, y2, inv, out, E);
}
